// MultiHeadAttention_11776800326311
// MI455X (gfx1250) — matrix operand order repaired and checked by reading, not compiled, not run on hardware
//
#include <hip/hip_runtime.h>
#include <math.h>

// ---------------------------------------------------------------------------
// MI455X (gfx1250) multi-head attention, bf16 WMMA pipeline.
//   B=2, S=4096, D=512, H=8, dh=64.  wave32, V_WMMA_F32_16X16X32_BF16.
//   GEMM tiles staged by the Tensor Data Mover (tensor_load_to_lds, TENSORcnt)
//   with double-buffered LDS so DMA overlaps WMMA compute.
// ---------------------------------------------------------------------------

typedef __bf16 bf16;
typedef __attribute__((ext_vector_type(16))) __bf16 v16bf;
typedef __attribute__((ext_vector_type(8)))  __bf16 bf16x8;   // 16 bytes
typedef __attribute__((ext_vector_type(8)))  float  v8f;
typedef __attribute__((ext_vector_type(4)))  unsigned int u32x4;
typedef __attribute__((ext_vector_type(8)))  int  i32x8;
typedef __attribute__((ext_vector_type(4)))  int  i32x4;

#define S_LEN   4096
#define D_DIM   512
#define H_NUM   8
#define DH      64
#define M_ROWS  8192            // B * S
#define QSCALE  0.18033688f     // (1/sqrt(64)) * log2(e)

#if defined(__has_builtin)
#  if __has_builtin(__builtin_amdgcn_tensor_load_to_lds) && \
      __has_builtin(__builtin_amdgcn_s_wait_tensorcnt)
#    define USE_TDM 1
#  endif
#endif
#ifndef USE_TDM
#  define USE_TDM 0
#endif

__device__ inline bf16 f2bf(float f) { return (bf16)f; }   // native v_cvt bf16

__device__ inline v16bf make_frag(bf16x8 lo, bf16x8 hi) {
    union { v16bf v; bf16x8 h[2]; } u;
    u.h[0] = lo; u.h[1] = hi;
    return u.v;
}

__device__ inline v8f wmma_bf16(v16bf a, v16bf b, v8f c) {
    // (neg_a, A, neg_b, B, c_mod, C, reuse_a, reuse_b)
    return __builtin_amdgcn_wmma_f32_16x16x32_bf16(false, a, false, b, (short)0, c,
                                                   false, false);
}

__device__ inline float bperm_f32(float v, int src_lane) {
    return __int_as_float(__builtin_amdgcn_ds_bpermute(src_lane << 2,
                                                       __float_as_int(v)));
}

#if USE_TDM
// ---------------------------------------------------------------------------
// Tensor Data Mover: 2D tile load (bf16 elements) into LDS.
// Descriptor layout per CDNA5 ISA (D# groups 0..3).
// All arguments are wave-uniform; tracked with TENSORcnt.
// ---------------------------------------------------------------------------
__device__ inline void tdm_load_tile_2d(unsigned lds_addr, const void* gptr,
                                        unsigned tile_w, unsigned tile_h,
                                        unsigned tensor_w, unsigned tensor_h,
                                        unsigned row_stride_elems) {
    unsigned long long ga = (unsigned long long)(uintptr_t)gptr;
    u32x4 g0;
    g0[0] = 1u;                                     // count=1, user-mode, no gather
    g0[1] = lds_addr;                               // LDS byte address
    g0[2] = (unsigned)(ga & 0xFFFFFFFFu);           // global_addr[31:0]
    g0[3] = (unsigned)((ga >> 32) & 0x01FFFFFFu)    // global_addr[56:32]
          | (2u << 30);                             // type = 2 ("image")
    i32x8 g1;
    g1[0] = (int)(1u << 16);                        // data_size=1 -> 2 bytes/elem
    g1[1] = (int)((tensor_w & 0xFFFFu) << 16);      // tensor_dim0[15:0]
    g1[2] = (int)(((tensor_w >> 16) & 0xFFFFu)      // tensor_dim0[31:16]
          | ((tensor_h & 0xFFFFu) << 16));          // tensor_dim1[15:0]
    g1[3] = (int)(((tensor_h >> 16) & 0xFFFFu)      // tensor_dim1[31:16]
          | ((tile_w & 0xFFFFu) << 16));            // tile_dim0
    g1[4] = (int)(tile_h & 0xFFFFu);                // tile_dim1 (tile_dim2 = 0)
    g1[5] = (int)row_stride_elems;                  // tensor_dim0_stride[31:0]
    g1[6] = 0;                                      // stride hi / dim1_stride lo
    g1[7] = 0;
    i32x4 gz = {0, 0, 0, 0};                        // groups 2/3 unused (2D)
#if __clang_major__ >= 23
    i32x8 gz8 = {0, 0, 0, 0, 0, 0, 0, 0};
    __builtin_amdgcn_tensor_load_to_lds(g0, g1, gz, gz, gz8, 0);
#else
    __builtin_amdgcn_tensor_load_to_lds(g0, g1, gz, gz, 0);
#endif
}
#endif

// ---------------------------------------------------------------------------
// Prep: fp32 -> bf16 conversion, and weight transpose [N,K] -> bf16 [K,N]
// ---------------------------------------------------------------------------
__global__ void cvt_kernel(const float* __restrict__ in, bf16* __restrict__ out, int n) {
    int i = blockIdx.x * blockDim.x + threadIdx.x;
    if (i < n) out[i] = f2bf(in[i]);
}

__global__ void wtr_kernel(const float* __restrict__ W, bf16* __restrict__ WT) {
    int i = blockIdx.x * blockDim.x + threadIdx.x;   // 512*512 threads
    int k = i >> 9, n = i & 511;
    WT[i] = f2bf(W[n * D_DIM + k]);                  // WT[k][n] = W[n][k]
}

// ---------------------------------------------------------------------------
// Shared WMMA micro-tile: wave computes 32x32 from LDS tiles.
// ---------------------------------------------------------------------------
__device__ inline void gemm_tile_compute(const bf16* Asb, const bf16* Bsb,
                                         int wm, int wn, int lane, int half, int l16,
                                         v8f (&acc)[2][2]) {
    v16bf af[2], bfg[2];
    #pragma unroll
    for (int mr = 0; mr < 2; ++mr) {
        int row = wm * 32 + mr * 16 + l16;
        af[mr] = make_frag(*(const bf16x8*)&Asb[row * 32 + half * 8],
                           *(const bf16x8*)&Asb[row * 32 + 16 + half * 8]);
    }
    #pragma unroll
    for (int nc = 0; nc < 2; ++nc) {
        int nb = wn * 32 + nc * 16;
        #pragma unroll
        for (int s = 0; s < 16; ++s)
            bfg[nc][s] = Bsb[((s < 8) ? half * 8 + s : 8 + half * 8 + s) * 64 + nb + l16];
    }
    #pragma unroll
    for (int mr = 0; mr < 2; ++mr)
        #pragma unroll
        for (int nc = 0; nc < 2; ++nc)
            acc[mr][nc] = wmma_bf16(af[mr], bfg[nc], acc[mr][nc]);
}

// ---------------------------------------------------------------------------
// Tiled GEMM:  Y[M,N] = X[M,512] * WT[512,N] + bias    (M=8192, N=512)
// block = 256 threads (8 waves), tile = 128(M) x 64(N), k-step 32.
// TDM double-buffered staging: tile k+1 DMA overlaps WMMA on tile k.
// mode 0: bf16 Q^T [B,H,dh,S] scaled | mode 1: bf16 [B,H,S,dh] | mode 2: f32 [M,N]
// ---------------------------------------------------------------------------
__global__ void gemm_kernel(const bf16* __restrict__ X, const bf16* __restrict__ WT,
                            const float* __restrict__ bias,
                            bf16* __restrict__ outb, float* __restrict__ outf,
                            int mode, float scale) {
    __shared__ alignas(16) bf16 As[2][128 * 32];   // 2 x 8 KB
    __shared__ alignas(16) bf16 Bs[2][32 * 64];    // 2 x 4 KB

    const int tid  = threadIdx.x;
    const int wv   = tid >> 5, lane = tid & 31;
    const int half = lane >> 4, l16 = lane & 15;
    const int wm   = wv >> 1,  wn   = wv & 1;
    const int n0   = blockIdx.x * 64;
    const int m0   = blockIdx.y * 128;

    const v8f vzero = {0.f, 0.f, 0.f, 0.f, 0.f, 0.f, 0.f, 0.f};
    v8f acc[2][2];
    acc[0][0] = vzero; acc[0][1] = vzero; acc[1][0] = vzero; acc[1][1] = vzero;

#if USE_TDM
    // -------- TDM double-buffered pipeline (wave 0 drives the DMA) --------
    if (wv == 0) {
        tdm_load_tile_2d((unsigned)(uintptr_t)&As[0][0], X + (m0 * D_DIM + 0),
                         32, 128, D_DIM, M_ROWS, D_DIM);
        tdm_load_tile_2d((unsigned)(uintptr_t)&Bs[0][0], WT + (0 * D_DIM + n0),
                         64, 32, D_DIM, D_DIM, D_DIM);
    }
    for (int kb = 0; kb < D_DIM; kb += 32) {
        const int cur = (kb >> 5) & 1;
        if (wv == 0) {
            if (kb + 32 < D_DIM) {
                tdm_load_tile_2d((unsigned)(uintptr_t)&As[cur ^ 1][0],
                                 X + (m0 * D_DIM + kb + 32),
                                 32, 128, D_DIM, M_ROWS, D_DIM);
                tdm_load_tile_2d((unsigned)(uintptr_t)&Bs[cur ^ 1][0],
                                 WT + ((kb + 32) * D_DIM + n0),
                                 64, 32, D_DIM, D_DIM, D_DIM);
                __builtin_amdgcn_s_wait_tensorcnt(2);   // older pair (cur) complete
            } else {
                __builtin_amdgcn_s_wait_tensorcnt(0);
            }
        }
        __syncthreads();                               // cur tiles visible to all
        gemm_tile_compute(&As[cur][0], &Bs[cur][0], wm, wn, lane, half, l16, acc);
        __syncthreads();                               // done reading before rewrite
    }
#else
    // -------- fallback: manual global->LDS staging, single buffer --------
    for (int kb = 0; kb < D_DIM; kb += 32) {
        __syncthreads();
        #pragma unroll
        for (int i = 0; i < 2; ++i) {
            int c = tid + i * 256;
            int row = c >> 2, cc = c & 3;
            *(bf16x8*)&As[0][row * 32 + cc * 8] =
                *(const bf16x8*)&X[(m0 + row) * D_DIM + kb + cc * 8];
        }
        {
            int row = tid >> 3, cc = tid & 7;
            *(bf16x8*)&Bs[0][row * 64 + cc * 8] =
                *(const bf16x8*)&WT[(kb + row) * D_DIM + n0 + cc * 8];
        }
        __syncthreads();
        gemm_tile_compute(&As[0][0], &Bs[0][0], wm, wn, lane, half, l16, acc);
    }
#endif

    // epilogue
    #pragma unroll
    for (int mr = 0; mr < 2; ++mr) {
        #pragma unroll
        for (int nc = 0; nc < 2; ++nc) {
            const int n     = n0 + wn * 32 + nc * 16 + l16;
            const float bv  = bias[n];
            const int mbase = m0 + wm * 32 + mr * 16 + 8 * half;  // 8 consecutive rows
            if (mode == 0) {
                // Q^T [B,H,dh,S], scaled: lane has fixed (h,d), rows contiguous in s
                int b = mbase >> 12, s = mbase & (S_LEN - 1);
                int hh = n >> 6, d = n & (DH - 1);
                bf16x8 v;
                #pragma unroll
                for (int r = 0; r < 8; ++r)
                    v[r] = f2bf((acc[mr][nc][r] + bv) * scale);
                *(bf16x8*)&outb[((size_t)((b * H_NUM + hh) * DH + d)) * S_LEN + s] = v;
            } else if (mode == 1) {
                // head layout [B,H,S,dh]
                int b = mbase >> 12, hh = n >> 6, d = n & (DH - 1);
                #pragma unroll
                for (int r = 0; r < 8; ++r) {
                    int s = (mbase + r) & (S_LEN - 1);
                    outb[((size_t)((b * H_NUM + hh) * S_LEN + s)) * DH + d] =
                        f2bf(acc[mr][nc][r] + bv);
                }
            } else {
                // f32 row-major final output
                #pragma unroll
                for (int r = 0; r < 8; ++r)
                    outf[(size_t)(mbase + r) * D_DIM + n] = acc[mr][nc][r] + bv;
            }
        }
    }
}

// ---------------------------------------------------------------------------
// Flash attention.  grid = (S/64, B*H), block = 128 (4 waves, 16 q each).
// scores^T = K_tile(A) x Q^T(B): C-frag (lane=q, vgpr=key) feeds directly as
// the A-frag of P x V — no LDS / shuffles for the P matrix at all.
// K/V stream straight from the 192 MB L2 (whole KV set is ~16 MB).
// ---------------------------------------------------------------------------
__global__ void attn_kernel(const bf16* __restrict__ QT, const bf16* __restrict__ Kh,
                            const bf16* __restrict__ Vh, bf16* __restrict__ ctx) {
    const int tid  = threadIdx.x;
    const int wv   = tid >> 5, lane = tid & 31;
    const int half = lane >> 4, l16 = lane & 15;
    const int bh   = blockIdx.y;
    const int b    = bh >> 3, h = bh & 7;
    const int q0   = blockIdx.x * 64 + wv * 16;

    const bf16* QTb = QT + (size_t)bh * DH * S_LEN;
    const bf16* Kb  = Kh + (size_t)bh * S_LEN * DH;
    const bf16* Vb  = Vh + (size_t)bh * S_LEN * DH;

    // Q^T B-fragments, resident for the whole pass (k-dim = dh = 2 x 32)
    v16bf qf[2];
    #pragma unroll
    for (int c = 0; c < 2; ++c) {
        #pragma unroll
        for (int s = 0; s < 16; ++s)
            qf[c][s] = QTb[(size_t)(c * 32 + ((s < 8) ? half * 8 + s : 8 + half * 8 + s)) * S_LEN + q0 + l16];
    }

    const v8f vzero = {0.f, 0.f, 0.f, 0.f, 0.f, 0.f, 0.f, 0.f};
    v8f accD[4];
    accD[0] = vzero; accD[1] = vzero; accD[2] = vzero; accD[3] = vzero;

    float mi = -INFINITY, li = 0.f;

    for (int key0 = 0; key0 < S_LEN; key0 += 32) {
        // prefetch next K/V rows into cache (global_prefetch_b8)
        int pk = (key0 + 32 < S_LEN) ? key0 + 32 : 0;
        __builtin_prefetch(Kb + (pk + l16) * DH, 0, 1);
        __builtin_prefetch(Vb + (pk + lane) * DH, 0, 1);

        // K A-fragments: 2 key tiles x 2 dh chunks
        v16bf ka[2][2];
        #pragma unroll
        for (int t = 0; t < 2; ++t)
            #pragma unroll
            for (int c = 0; c < 2; ++c) {
                const bf16* p = Kb + (key0 + t * 16 + l16) * DH + c * 32 + half * 8;
                ka[t][c] = make_frag(*(const bf16x8*)p, *(const bf16x8*)(p + 16));
            }

        // scores^T for 32 keys (f32 accum)
        v8f s0 = vzero, s1 = vzero;
        s0 = wmma_bf16(ka[0][0], qf[0], s0);
        s0 = wmma_bf16(ka[0][1], qf[1], s0);
        s1 = wmma_bf16(ka[1][0], qf[0], s1);
        s1 = wmma_bf16(ka[1][1], qf[1], s1);

        // online softmax (per-lane q = l16; halves combined via shfl_xor 16)
        float mloc = mi;
        #pragma unroll
        for (int r = 0; r < 8; ++r)
            mloc = fmaxf(mloc, fmaxf(s0[r], s1[r]));
        mloc = fmaxf(mloc, __shfl_xor(mloc, 16, 32));

        float rescale = exp2f(mi - mloc);
        float p0[8], p1[8], lsum = 0.f;
        #pragma unroll
        for (int r = 0; r < 8; ++r) {
            p0[r] = exp2f(s0[r] - mloc);
            p1[r] = exp2f(s1[r] - mloc);
            lsum += p0[r] + p1[r];
        }
        lsum += __shfl_xor(lsum, 16, 32);
        li = li * rescale + lsum;
        mi = mloc;

        // rescale ctx accumulators: row r of acc is q' = 8*half + r; its factor
        // lives at lane (8*half + r) -> ds_bpermute broadcast
        #pragma unroll
        for (int r = 0; r < 8; ++r) {
            float sc = bperm_f32(rescale, half * 8 + r);
            accD[0][r] *= sc; accD[1][r] *= sc; accD[2][r] *= sc; accD[3][r] *= sc;
        }

        // P A-fragment: element e<8 -> key 8*half+e (tile0), e>=8 -> tile1
        v16bf pf;
        #pragma unroll
        for (int e = 0; e < 8; ++e) {
            pf[e]     = f2bf(p0[e]);
            pf[e + 8] = f2bf(p1[e]);
        }

        // ctx += P x V  (V B fragment: lane column d = l16, elements over keys in table order)
        #pragma unroll
        for (int dt = 0; dt < 4; ++dt) {
            v16bf vf;
            #pragma unroll
            for (int s = 0; s < 16; ++s)
                vf[s] = Vb[(key0 + ((s < 8) ? half * 8 + s : 8 + half * 8 + s)) * DH + dt * 16 + l16];
            accD[dt] = wmma_bf16(pf, vf, accD[dt]);
        }
    }

    // finalize: divide row q' = 8*half + r by its li, write bf16 concat [B,S,D]
    float linv[8];
    #pragma unroll
    for (int r = 0; r < 8; ++r) {
        float lq = bperm_f32(li, half * 8 + r);
        linv[r] = 1.f / lq;
    }
    #pragma unroll
    for (int dt = 0; dt < 4; ++dt) {
        int d = h * DH + dt * 16 + l16;
        #pragma unroll
        for (int r = 0; r < 8; ++r) {
            int s = q0 + 8 * half + r;
            ctx[((size_t)(b * S_LEN + s)) * D_DIM + d] = f2bf(accD[dt][r] * linv[r]);
        }
    }
}

// ---------------------------------------------------------------------------
// Host-side orchestration
// ---------------------------------------------------------------------------
extern "C" void kernel_launch(void* const* d_in, const int* in_sizes, int n_in,
                              void* d_out, int out_size, void* d_ws, size_t ws_size,
                              hipStream_t stream) {
    const float* q  = (const float*)d_in[0];
    const float* k  = (const float*)d_in[1];
    const float* v  = (const float*)d_in[2];
    const float* Wq = (const float*)d_in[3];
    const float* bq = (const float*)d_in[4];
    const float* Wk = (const float*)d_in[5];
    const float* bk = (const float*)d_in[6];
    const float* Wv = (const float*)d_in[7];
    const float* bv = (const float*)d_in[8];
    const float* Wo = (const float*)d_in[9];
    const float* bo = (const float*)d_in[10];

    const size_t SZ_X = (size_t)M_ROWS * D_DIM * sizeof(bf16);   // 8 MB
    const size_t SZ_W = (size_t)D_DIM * D_DIM * sizeof(bf16);    // 0.5 MB

    char* ws = (char*)d_ws;
    bf16* Xq  = (bf16*)(ws);
    bf16* Xk  = (bf16*)(ws + SZ_X);
    bf16* Xv  = (bf16*)(ws + 2 * SZ_X);
    bf16* WqT = (bf16*)(ws + 3 * SZ_X);
    bf16* WkT = (bf16*)(ws + 3 * SZ_X + SZ_W);
    bf16* WvT = (bf16*)(ws + 3 * SZ_X + 2 * SZ_W);
    bf16* WoT = (bf16*)(ws + 3 * SZ_X + 3 * SZ_W);
    bf16* QTh = (bf16*)(ws + 3 * SZ_X + 4 * SZ_W);               // [B,H,dh,S]
    bf16* Khd = (bf16*)(ws + 4 * SZ_X + 4 * SZ_W);               // [B,H,S,dh]
    bf16* Vhd = (bf16*)(ws + 5 * SZ_X + 4 * SZ_W);               // [B,H,S,dh]
    bf16* CTX = (bf16*)(ws + 6 * SZ_X + 4 * SZ_W);               // [B,S,D]

    const int NX = M_ROWS * D_DIM;                               // 4,194,304
    cvt_kernel<<<NX / 256, 256, 0, stream>>>(q, Xq, NX);
    cvt_kernel<<<NX / 256, 256, 0, stream>>>(k, Xk, NX);
    cvt_kernel<<<NX / 256, 256, 0, stream>>>(v, Xv, NX);

    const int NW = D_DIM * D_DIM;                                // 262,144
    wtr_kernel<<<NW / 256, 256, 0, stream>>>(Wq, WqT);
    wtr_kernel<<<NW / 256, 256, 0, stream>>>(Wk, WkT);
    wtr_kernel<<<NW / 256, 256, 0, stream>>>(Wv, WvT);
    wtr_kernel<<<NW / 256, 256, 0, stream>>>(Wo, WoT);

    dim3 ggrid(D_DIM / 64, M_ROWS / 128);                        // (8, 64)
    gemm_kernel<<<ggrid, 256, 0, stream>>>(Xq, WqT, bq, QTh, nullptr, 0, QSCALE);
    gemm_kernel<<<ggrid, 256, 0, stream>>>(Xk, WkT, bk, Khd, nullptr, 1, 1.f);
    gemm_kernel<<<ggrid, 256, 0, stream>>>(Xv, WvT, bv, Vhd, nullptr, 1, 1.f);

    dim3 agrid(S_LEN / 64, 2 * H_NUM);                           // (64, 16)
    attn_kernel<<<agrid, 128, 0, stream>>>(QTh, Khd, Vhd, CTX);

    gemm_kernel<<<ggrid, 256, 0, stream>>>(CTX, WoT, bo, nullptr, (float*)d_out, 2, 1.f);
}
